// QuantumGeometricAttention_33328946217773
// MI455X (gfx1250) — compile-verified, matrix operand order checked by reading, not run on hardware
//
#include <hip/hip_runtime.h>

// ---------------------------------------------------------------------------
// CDNA5 (gfx1250) wave32 WMMA implementation of QuantumGeometricAttention.
// All heavy GEMMs run on v_wmma_f32_16x16x32_bf16; energy/softmax in f32 VALU.
// BF16 WMMA has no A/B negation (NEG = {CNeg,0,0}), so the complex-multiply
// subtraction uses a precomputed negated-imaginary weight plane instead.
// ---------------------------------------------------------------------------

typedef __bf16 bf16_t;
typedef __attribute__((ext_vector_type(16))) __bf16 v16bf;
typedef __attribute__((ext_vector_type(8)))  __bf16 v8bf;
typedef __attribute__((ext_vector_type(8)))  float  v8f;

#define WMMA_BF16(Av, Bv, Cv) \
  __builtin_amdgcn_wmma_f32_16x16x32_bf16(false, (Av), false, (Bv), (short)0, (Cv), false, false)

// Problem constants (from the reference): B=4,S=1024 -> M=4096; Hd=Dm=512; H=8; L=3; TILE=8
constexpr int MROWS = 4096;
constexpr int DIM   = 512;     // Hd == Dm == 512
constexpr int NHEAD = 8;
constexpr int NLAYER = 3;

// ---------------------------------------------------------------------------
// Fragment loader for 16-bit A/B matrices (16x32 / 32x16), CDNA5 wave32 layout:
//   lane = half*16 + lr ;  row/col = lr ;
//   elements 0..7  -> K = half*8 + 0..7
//   elements 8..15 -> K = half*8 + 16..23
// Each lane does two 16-byte loads (global_load_b128 / ds_load_b128).
// ---------------------------------------------------------------------------
__device__ __forceinline__ v16bf load_frag(const bf16_t* __restrict__ plane,
                                           int ld, int rc_base, int k0) {
  const int lane = threadIdx.x & 31;
  const int half = lane >> 4;
  const int lr   = lane & 15;
  const bf16_t* p = plane + (size_t)(rc_base + lr) * ld + (k0 + half * 8);
  union { v16bf v; v8bf h[2]; } u;
  u.h[0] = *(const v8bf*)(p);
  u.h[1] = *(const v8bf*)(p + 16);
  return u.v;
}

// ---------------------------------------------------------------------------
// zero helper
// ---------------------------------------------------------------------------
__global__ void zero_kernel(float* __restrict__ p, int n) {
  int i = blockIdx.x * blockDim.x + threadIdx.x;
  if (i < n) p[i] = 0.f;
}

// complex64 interleaved -> bf16 planes: re, im, and negated im (for cre WMMA)
__global__ void cvt_w_kernel(const float2* __restrict__ w,
                             bf16_t* __restrict__ re, bf16_t* __restrict__ im,
                             bf16_t* __restrict__ imn, int n) {
  int i = blockIdx.x * blockDim.x + threadIdx.x;
  if (i < n) {
    float2 v = w[i];
    re[i]  = (bf16_t)v.x;
    im[i]  = (bf16_t)v.y;
    imn[i] = (bf16_t)(-v.y);
  }
}

// pack x (separate f32 re/im) into bf16 planes + per-row |x|^2 energy
__global__ __launch_bounds__(256) void pack_x_kernel(
    const float* __restrict__ xr, const float* __restrict__ xi,
    bf16_t* __restrict__ Are, bf16_t* __restrict__ Aim, float* __restrict__ e_in) {
  const int row = blockIdx.x;
  const size_t base = (size_t)row * DIM;
  float e = 0.f;
  for (int i = threadIdx.x; i < DIM; i += blockDim.x) {
    float r = xr[base + i], m = xi[base + i];
    Are[base + i] = (bf16_t)r;
    Aim[base + i] = (bf16_t)m;
    e += r * r + m * m;
  }
  for (int o = 1; o < 32; o <<= 1) e += __shfl_xor(e, o, 32);
  __shared__ float red[8];
  if ((threadIdx.x & 31) == 0) red[threadIdx.x >> 5] = e;
  __syncthreads();
  if (threadIdx.x == 0) {
    float s = 0.f;
    for (int wv = 0; wv < 8; ++wv) s += red[wv];
    e_in[row] = s;
  }
}

// ---------------------------------------------------------------------------
// Generic complex GEMM:  C[M,N] = A[M,K] * W^T  (+bias)   (y[o] = sum_d x[d]W[o,d])
// A given as bf16 re/im planes (row-major M x K).
// W given as bf16 re/im/(-im) planes (row-major N x K) -> B[k][n] = W[n*K + k],
// so a B-fragment is a contiguous 16B run of k for a fixed n: perfect layout fit.
//   cre = are*bre + aim*(-bim)      (negated plane replaces unsupported neg_a)
//   cim = are*bim + aim*bre
// Block tile 128(M) x 64(N), 8 waves (2 along M x 4 along N), 4 tile-pairs/wave.
// ---------------------------------------------------------------------------
template<bool ADD_BIAS, bool ENERGY, bool OUT_F32, bool OUT_BF16>
__global__ __launch_bounds__(256) void cgemm_kernel(
    const bf16_t* __restrict__ Are, const bf16_t* __restrict__ Aim,
    const bf16_t* __restrict__ Wre, const bf16_t* __restrict__ Wim,
    const bf16_t* __restrict__ Wimn,
    const float2* __restrict__ bias,
    float*  __restrict__ Cre,  float*  __restrict__ Cim,
    bf16_t* __restrict__ Obre, bf16_t* __restrict__ Obim,
    float*  __restrict__ e_out) {
  constexpr int K = DIM, N = DIM;
  const int wid  = threadIdx.x >> 5;
  const int lane = threadIdx.x & 31;
  const int half = lane >> 4, lr = lane & 15;
  const int m0 = blockIdx.x * 128 + (wid >> 2) * 64;   // wave's 64-row strip
  const int n  = blockIdx.y * 64  + (wid & 3) * 16;    // wave's 16-col tile

  const v8f z = {0.f,0.f,0.f,0.f,0.f,0.f,0.f,0.f};
  v8f cre[4], cim[4];
  for (int t = 0; t < 4; ++t) { cre[t] = z; cim[t] = z; }

  for (int k0 = 0; k0 < K; k0 += 32) {
    if (k0 + 32 < K) {
      __builtin_prefetch(Wre + (size_t)n * K + (k0 + 32), 0, 1);
      __builtin_prefetch(Are + (size_t)m0 * K + (k0 + 32), 0, 1);
    }
    const v16bf bre  = load_frag(Wre,  K, n, k0);
    const v16bf bim  = load_frag(Wim,  K, n, k0);
    const v16bf bimn = load_frag(Wimn, K, n, k0);
#pragma unroll
    for (int t = 0; t < 4; ++t) {
      const v16bf are = load_frag(Are, K, m0 + 16 * t, k0);
      const v16bf aim = load_frag(Aim, K, m0 + 16 * t, k0);
      cre[t] = WMMA_BF16(are, bre,  cre[t]);   // + re*re
      cre[t] = WMMA_BF16(aim, bimn, cre[t]);   // - im*im (negated plane)
      cim[t] = WMMA_BF16(are, bim,  cim[t]);   // + re*im
      cim[t] = WMMA_BF16(aim, bre,  cim[t]);   // + im*re
    }
  }

  float2 bv = make_float2(0.f, 0.f);
  if (ADD_BIAS) bv = bias[n + lr];
#pragma unroll
  for (int t = 0; t < 4; ++t) {
#pragma unroll
    for (int j = 0; j < 8; ++j) {
      const int row = m0 + 16 * t + j + 8 * half;   // C-layout: M = j + 8*(lane/16)
      const int col = n + lr;                       //           N = lane%16
      float re = cre[t][j], im = cim[t][j];
      if (ADD_BIAS) { re += bv.x; im += bv.y; }
      const size_t idx = (size_t)row * N + col;
      if (OUT_F32)  { Cre[idx] = re;            Cim[idx] = im; }
      if (OUT_BF16) { Obre[idx] = (bf16_t)re;   Obim[idx] = (bf16_t)im; }
      if (ENERGY) {
        float e = re * re + im * im;
        for (int o = 1; o < 16; o <<= 1) e += __shfl_xor(e, o, 32); // reduce over 16 cols
        if (lr == 0) atomicAdd(&e_out[row], e);
      }
    }
  }
}

// energy-conserving rescale; optionally emit bf16 planes and/or interleaved complex f32
__global__ void ec_scale_kernel(
    const float* __restrict__ Cre, const float* __restrict__ Cim,
    const float* __restrict__ e_in, const float* __restrict__ e_out,
    bf16_t* __restrict__ Obre, bf16_t* __restrict__ Obim,
    float2* __restrict__ Ocplx, int total) {
  int i = blockIdx.x * blockDim.x + threadIdx.x;
  if (i >= total) return;
  const int row = i / DIM;
  const float s = sqrtf(e_in[row] / (e_out[row] + 1e-8f));
  const float re = Cre[i] * s, im = Cim[i] * s;
  if (Obre)  { Obre[i] = (bf16_t)re; Obim[i] = (bf16_t)im; }
  if (Ocplx) { Ocplx[i] = make_float2(re, im); }
}

// ---------------------------------------------------------------------------
// Fused tiled motivic attention.
// One workgroup (8 waves) per 16-row pair of 8-row tiles; loops all 8 heads.
//   q = m*Wq^T, k = m*Wk^T           (WMMA, per-wave 64-col strip)
//   scores = Re(q k^H)/sqrt(D)       (WMMA: qre*kre^T + qim*kim^T via LDS restage)
//   softmax over 8x8 diagonal blocks
//   out += (attn*m)*Wv^T             (attn*m cheap in VALU; WMMA GEMM vs Wv)
// Output: mean over heads as bf16 planes + per-row energy (e_in of inverse proj).
// LDS: m(32KB) + q,k(64KB, aliased by attn*m) + scores(1KB) ~= 99KB of 320KB/WGP.
// ---------------------------------------------------------------------------
__global__ __launch_bounds__(256) void tile_attn_kernel(
    const bf16_t* __restrict__ Mre, const bf16_t* __restrict__ Mim,
    const bf16_t* __restrict__ Wqre, const bf16_t* __restrict__ Wqim, const bf16_t* __restrict__ Wqimn,
    const bf16_t* __restrict__ Wkre, const bf16_t* __restrict__ Wkim, const bf16_t* __restrict__ Wkimn,
    const bf16_t* __restrict__ Wvre, const bf16_t* __restrict__ Wvim, const bf16_t* __restrict__ Wvimn,
    bf16_t* __restrict__ Ore, bf16_t* __restrict__ Oim,
    float* __restrict__ e_in2) {
  extern __shared__ char smem[];
  bf16_t* m_re = (bf16_t*)smem;            // [16][DIM]
  bf16_t* m_im = m_re + 16 * DIM;
  bf16_t* q_re = m_im + 16 * DIM;
  bf16_t* q_im = q_re + 16 * DIM;
  bf16_t* k_re = q_im + 16 * DIM;
  bf16_t* k_im = k_re + 16 * DIM;
  float*  sc   = (float*)(k_im + 16 * DIM); // [16][16]
  bf16_t* am_re = q_re;  // alias: attn*m overwrites q after scores are done
  bf16_t* am_im = q_im;

  const int tid  = threadIdx.x;
  const int wid  = tid >> 5;
  const int lane = tid & 31;
  const int half = lane >> 4, lr = lane & 15;
  const int r0    = blockIdx.x * 16;   // two 8-row tiles
  const int ncol0 = wid * 64;          // wave's 64-col output strip
  const float inv_sqrt_d = rsqrtf((float)DIM);

  // stage the 16 x DIM m-tile into LDS (16B vectors)
  {
    const uint4* sr = (const uint4*)(Mre + (size_t)r0 * DIM);
    const uint4* si = (const uint4*)(Mim + (size_t)r0 * DIM);
    uint4* dr = (uint4*)m_re; uint4* di = (uint4*)m_im;
    const int nvec = 16 * DIM / 8;
    for (int i = tid; i < nvec; i += 256) { dr[i] = sr[i]; di[i] = si[i]; }
  }
  const v8f z = {0.f,0.f,0.f,0.f,0.f,0.f,0.f,0.f};
  v8f ore[4], oim[4];
  for (int t = 0; t < 4; ++t) { ore[t] = z; oim[t] = z; }
  __syncthreads();

  for (int h = 0; h < NHEAD; ++h) {
    const size_t woff = (size_t)h * DIM * DIM;
    v8f qre[4], qim[4], kre[4], kim[4];
    for (int t = 0; t < 4; ++t) { qre[t] = z; qim[t] = z; kre[t] = z; kim[t] = z; }

    // q,k projections for this head
    for (int k0 = 0; k0 < DIM; k0 += 32) {
      const v16bf are = load_frag(m_re, DIM, 0, k0);   // ds_load_b128
      const v16bf aim = load_frag(m_im, DIM, 0, k0);
#pragma unroll
      for (int t = 0; t < 4; ++t) {
        const int n = ncol0 + 16 * t;
        const v16bf bqr  = load_frag(Wqre  + woff, DIM, n, k0);
        const v16bf bqi  = load_frag(Wqim  + woff, DIM, n, k0);
        const v16bf bqin = load_frag(Wqimn + woff, DIM, n, k0);
        qre[t] = WMMA_BF16(are, bqr,  qre[t]);
        qre[t] = WMMA_BF16(aim, bqin, qre[t]);
        qim[t] = WMMA_BF16(are, bqi,  qim[t]);
        qim[t] = WMMA_BF16(aim, bqr,  qim[t]);
        const v16bf bkr  = load_frag(Wkre  + woff, DIM, n, k0);
        const v16bf bki  = load_frag(Wkim  + woff, DIM, n, k0);
        const v16bf bkin = load_frag(Wkimn + woff, DIM, n, k0);
        kre[t] = WMMA_BF16(are, bkr,  kre[t]);
        kre[t] = WMMA_BF16(aim, bkin, kre[t]);
        kim[t] = WMMA_BF16(are, bki,  kim[t]);
        kim[t] = WMMA_BF16(aim, bkr,  kim[t]);
      }
    }

    // restage q,k to LDS (bf16) and zero scores
    sc[tid] = 0.f;   // 256 == 16*16 entries
#pragma unroll
    for (int t = 0; t < 4; ++t) {
      const int colb = ncol0 + 16 * t + lr;
#pragma unroll
      for (int j = 0; j < 8; ++j) {
        const int rloc = j + 8 * half;
        q_re[rloc * DIM + colb] = (bf16_t)qre[t][j];
        q_im[rloc * DIM + colb] = (bf16_t)qim[t][j];
        k_re[rloc * DIM + colb] = (bf16_t)kre[t][j];
        k_im[rloc * DIM + colb] = (bf16_t)kim[t][j];
      }
    }
    __syncthreads();

    // partial scores over this wave's 64 cols: Re(q k^H) = qre*kre^T + qim*kim^T
    v8f s_acc = z;
#pragma unroll
    for (int st = 0; st < 2; ++st) {
      const int e0 = ncol0 + st * 32;
      const v16bf aqr = load_frag(q_re, DIM, 0, e0);
      const v16bf aqi = load_frag(q_im, DIM, 0, e0);
      const v16bf bkr = load_frag(k_re, DIM, 0, e0);  // B-frag: n = row s of k
      const v16bf bki = load_frag(k_im, DIM, 0, e0);
      s_acc = WMMA_BF16(aqr, bkr, s_acc);
      s_acc = WMMA_BF16(aqi, bki, s_acc);
    }
#pragma unroll
    for (int j = 0; j < 8; ++j)
      atomicAdd(&sc[(j + 8 * half) * 16 + lr], s_acc[j]);   // ds_add_f32
    __syncthreads();

    // softmax on the two 8x8 diagonal blocks (one thread per row)
    if (tid < 16) {
      const int bs = (tid < 8) ? 0 : 8;
      float v[8]; float mx = -3.4e38f;
#pragma unroll
      for (int s2 = 0; s2 < 8; ++s2) {
        v[s2] = sc[tid * 16 + bs + s2] * inv_sqrt_d;
        mx = fmaxf(mx, v[s2]);
      }
      float sum = 0.f;
#pragma unroll
      for (int s2 = 0; s2 < 8; ++s2) { v[s2] = __expf(v[s2] - mx); sum += v[s2]; }
      const float inv = 1.f / sum;
#pragma unroll
      for (int s2 = 0; s2 < 8; ++s2) sc[tid * 16 + bs + s2] = v[s2] * inv;
    }
    __syncthreads();

    // am = attn * m (block-diagonal, 8-wide), write into q region as bf16
    for (int it = 0; it < (16 * DIM) / 256; ++it) {
      const int flat = it * 256 + tid;
      const int t = flat / DIM;
      const int e = flat % DIM;
      const int bs = (t < 8) ? 0 : 8;
      float ar = 0.f, ai = 0.f;
#pragma unroll
      for (int s2 = 0; s2 < 8; ++s2) {
        const float a = sc[t * 16 + bs + s2];
        ar += a * (float)m_re[(bs + s2) * DIM + e];
        ai += a * (float)m_im[(bs + s2) * DIM + e];
      }
      am_re[flat] = (bf16_t)ar;
      am_im[flat] = (bf16_t)ai;
    }
    __syncthreads();

    // out += am * Wv^T   (head accumulation; mean applied at the end)
    for (int k0 = 0; k0 < DIM; k0 += 32) {
      const v16bf are = load_frag(am_re, DIM, 0, k0);
      const v16bf aim = load_frag(am_im, DIM, 0, k0);
#pragma unroll
      for (int t = 0; t < 4; ++t) {
        const int n = ncol0 + 16 * t;
        const v16bf bvr  = load_frag(Wvre  + woff, DIM, n, k0);
        const v16bf bvi  = load_frag(Wvim  + woff, DIM, n, k0);
        const v16bf bvin = load_frag(Wvimn + woff, DIM, n, k0);
        ore[t] = WMMA_BF16(are, bvr,  ore[t]);
        ore[t] = WMMA_BF16(aim, bvin, ore[t]);
        oim[t] = WMMA_BF16(are, bvi,  oim[t]);
        oim[t] = WMMA_BF16(aim, bvr,  oim[t]);
      }
    }
    __syncthreads();
  }

  // epilogue: mean over heads, bf16 planes for the inverse projection, row energy
  const float invH = 1.f / (float)NHEAD;
#pragma unroll
  for (int t = 0; t < 4; ++t) {
    const int col = ncol0 + 16 * t + lr;
#pragma unroll
    for (int j = 0; j < 8; ++j) {
      const int row = r0 + j + 8 * half;
      const float re = ore[t][j] * invH;
      const float im = oim[t][j] * invH;
      const size_t idx = (size_t)row * DIM + col;
      Ore[idx] = (bf16_t)re;
      Oim[idx] = (bf16_t)im;
      float e = re * re + im * im;
      for (int o = 1; o < 16; o <<= 1) e += __shfl_xor(e, o, 32);
      if (lr == 0) atomicAdd(&e_in2[row], e);
    }
  }
}

// ---------------------------------------------------------------------------
// Host-side orchestration (single stream, graph-capture safe: no alloc/sync).
// ---------------------------------------------------------------------------
extern "C" void kernel_launch(void* const* d_in, const int* in_sizes, int n_in,
                              void* d_out, int out_size, void* d_ws, size_t ws_size,
                              hipStream_t stream) {
  const float*  xr  = (const float*)d_in[0];
  const float*  xi  = (const float*)d_in[1];
  const float2* Wp  = (const float2*)d_in[2];
  const float2* bp  = (const float2*)d_in[3];
  const float2* WLw = (const float2*)d_in[4];
  const float2* bLw = (const float2*)d_in[5];
  const float2* Wq  = (const float2*)d_in[6];
  const float2* Wk  = (const float2*)d_in[7];
  const float2* Wv  = (const float2*)d_in[8];
  const float2* Wpi = (const float2*)d_in[9];
  const float2* bpi = (const float2*)d_in[10];

  const size_t MK = (size_t)MROWS * DIM;       // 4096*512
  const size_t WW = (size_t)DIM * DIM;         // 512*512

  unsigned char* base = (unsigned char*)d_ws;
  size_t off = 0;
  auto alloc = [&](size_t bytes) -> void* {
    void* p = base + off;
    off += (bytes + 255) & ~(size_t)255;
    return p;
  };
  float*  e_in1  = (float*)alloc(MROWS * 4);
  float*  e_out1 = (float*)alloc(MROWS * 4);   // e_out1,e_in2,e_out2 contiguous
  float*  e_in2  = (float*)alloc(MROWS * 4);
  float*  e_out2 = (float*)alloc(MROWS * 4);
  bf16_t* a0r = (bf16_t*)alloc(MK * 2);
  bf16_t* a0i = (bf16_t*)alloc(MK * 2);
  bf16_t* a1r = (bf16_t*)alloc(MK * 2);
  bf16_t* a1i = (bf16_t*)alloc(MK * 2);
  float*  cfr = (float*)alloc(MK * 4);
  float*  cfi = (float*)alloc(MK * 4);
  bf16_t* wpr = (bf16_t*)alloc(WW * 2);
  bf16_t* wpi_ = (bf16_t*)alloc(WW * 2);
  bf16_t* wpn = (bf16_t*)alloc(WW * 2);
  bf16_t* wlr = (bf16_t*)alloc(NLAYER * WW * 2);
  bf16_t* wli = (bf16_t*)alloc(NLAYER * WW * 2);
  bf16_t* wln = (bf16_t*)alloc(NLAYER * WW * 2);
  bf16_t* wqr = (bf16_t*)alloc(NHEAD * WW * 2);
  bf16_t* wqi = (bf16_t*)alloc(NHEAD * WW * 2);
  bf16_t* wqn = (bf16_t*)alloc(NHEAD * WW * 2);
  bf16_t* wkr = (bf16_t*)alloc(NHEAD * WW * 2);
  bf16_t* wki = (bf16_t*)alloc(NHEAD * WW * 2);
  bf16_t* wkn = (bf16_t*)alloc(NHEAD * WW * 2);
  bf16_t* wvr = (bf16_t*)alloc(NHEAD * WW * 2);
  bf16_t* wvi = (bf16_t*)alloc(NHEAD * WW * 2);
  bf16_t* wvn = (bf16_t*)alloc(NHEAD * WW * 2);
  bf16_t* wir = (bf16_t*)alloc(WW * 2);
  bf16_t* wii = (bf16_t*)alloc(WW * 2);
  bf16_t* win = (bf16_t*)alloc(WW * 2);
  (void)ws_size; (void)n_in; (void)in_sizes; (void)out_size;

  // 1) zero the atomic energy accumulators (e_out1,e_in2,e_out2 contiguous)
  zero_kernel<<<(3 * MROWS + 255) / 256, 256, 0, stream>>>(e_out1, 3 * MROWS);

  // 2) weight conversion f32-complex -> bf16 planes (re, im, -im)
  auto cvt = [&](const float2* w, bf16_t* re, bf16_t* im, bf16_t* imn, size_t n) {
    cvt_w_kernel<<<(unsigned)((n + 255) / 256), 256, 0, stream>>>(w, re, im, imn, (int)n);
  };
  cvt(Wp,  wpr, wpi_, wpn, WW);
  cvt(WLw, wlr, wli,  wln, NLAYER * WW);
  cvt(Wq,  wqr, wqi,  wqn, NHEAD * WW);
  cvt(Wk,  wkr, wki,  wkn, NHEAD * WW);
  cvt(Wv,  wvr, wvi,  wvn, NHEAD * WW);
  cvt(Wpi, wir, wii,  win, WW);

  // 3) pack x + per-row input energy
  pack_x_kernel<<<MROWS, 256, 0, stream>>>(xr, xi, a0r, a0i, e_in1);

  const dim3 gemm_grid(MROWS / 128, DIM / 64);

  // 4) EC projection: C = x * Wp^T (+bp) with output energy  (f32 out)
  cgemm_kernel<true, true, true, false><<<gemm_grid, 256, 0, stream>>>(
      a0r, a0i, wpr, wpi_, wpn, bp, cfr, cfi, nullptr, nullptr, e_out1);

  // 5) energy-conserving rescale -> bf16 activation planes (act0)
  ec_scale_kernel<<<(unsigned)((MK + 255) / 256), 256, 0, stream>>>(
      cfr, cfi, e_in1, e_out1, a0r, a0i, nullptr, (int)MK);

  // 6) three chained complex layers (ping-pong act0 <-> act1)
  bf16_t* srcs[2] = {a0r, a1r};
  bf16_t* srci[2] = {a0i, a1i};
  for (int l = 0; l < NLAYER; ++l) {
    const int s = l & 1, d = s ^ 1;
    cgemm_kernel<true, false, false, true><<<gemm_grid, 256, 0, stream>>>(
        srcs[s], srci[s], wlr + (size_t)l * WW, wli + (size_t)l * WW, wln + (size_t)l * WW,
        bLw + (size_t)l * DIM, nullptr, nullptr, srcs[d], srci[d], nullptr);
  }
  // after 3 layers the live activation is in act1
  bf16_t* mr = a1r; bf16_t* mi = a1i;
  bf16_t* outr = a0r; bf16_t* outi = a0i;

  // 7) fused per-tile multi-head attention (LDS ~99KB of the 320KB WGP pool)
  const size_t smem_bytes = (size_t)6 * 16 * DIM * sizeof(bf16_t) + 16 * 16 * sizeof(float);
  tile_attn_kernel<<<MROWS / 16, 256, smem_bytes, stream>>>(
      mr, mi, wqr, wqi, wqn, wkr, wki, wkn, wvr, wvi, wvn, outr, outi, e_in2);

  // 8) EC inverse projection: C = out * Wpi^T (+bpi) with output energy
  cgemm_kernel<true, true, true, false><<<gemm_grid, 256, 0, stream>>>(
      outr, outi, wir, wii, win, bpi, cfr, cfi, nullptr, nullptr, e_out2);

  // 9) final energy-conserving rescale -> interleaved complex64 output
  ec_scale_kernel<<<(unsigned)((MK + 255) / 256), 256, 0, stream>>>(
      cfr, cfi, e_in2, e_out2, nullptr, nullptr, (float2*)d_out, (int)MK);
}
